// SparseAttention_14044543057928
// MI455X (gfx1250) — matrix operand order repaired and checked by reading, not compiled, not run on hardware
//
#include <hip/hip_runtime.h>
#include <hip/hip_bf16.h>

typedef __bf16 bf16;
typedef __attribute__((ext_vector_type(16))) __bf16 v16bf;
typedef __attribute__((ext_vector_type(4)))  __bf16 v4bf;
typedef __attribute__((ext_vector_type(8)))  float   v8f;
typedef __attribute__((ext_vector_type(4)))  float   v4f;
typedef __attribute__((ext_vector_type(4)))  int     v4i;

#define WMMA_BF16(a, b, c) \
  __builtin_amdgcn_wmma_f32_16x16x32_bf16(false, (a), false, (b), (short)0, (c), false, false)

// ---- CDNA5 async global->LDS copy (ASYNCcnt-tracked), with sync fallback ----
#if defined(__has_builtin)
#if __has_builtin(__builtin_amdgcn_global_load_async_to_lds_b128)
#define HAVE_ASYNC_LDS 1
#endif
#endif
#ifndef HAVE_ASYNC_LDS
#define HAVE_ASYNC_LDS 0
#endif

#if HAVE_ASYNC_LDS
typedef __attribute__((address_space(1))) v4i* gv4i_p;  // <4 x i32> in global (AS1)
typedef __attribute__((address_space(3))) v4i* lv4i_p;  // <4 x i32> in LDS (AS3)
#if __has_builtin(__builtin_amdgcn_s_wait_asynccnt)
#define WAIT_ASYNC() __builtin_amdgcn_s_wait_asynccnt(0)
#else
#define WAIT_ASYNC() asm volatile("s_wait_asynccnt 0x0" ::: "memory")
#endif
#else
#define WAIT_ASYNC() ((void)0)
#endif

__device__ __forceinline__ void copy16_to_lds(const bf16* g, bf16* l) {
#if HAVE_ASYNC_LDS
  gv4i_p gp = (gv4i_p)(unsigned long long)g;                    // flat == global addr
  lv4i_p lp = (lv4i_p)(unsigned int)(unsigned long long)l;      // low 32 bits == LDS offset
  __builtin_amdgcn_global_load_async_to_lds_b128(gp, lp, 0, 0);
#else
  *(v4f*)l = *(const v4f*)g;
#endif
}

static constexpr int B_  = 2;
static constexpr int S_  = 2048;
static constexpr int D_  = 1024;
static constexpr int H_  = 16;
static constexpr int HD_ = 64;
static constexpr int M_  = B_ * S_;   // 4096
static constexpr int TOPK_ = 3;

// ---------------------------------------------------------------- convert
__global__ __launch_bounds__(256) void f32_to_bf16_kernel(
    const float* __restrict__ in, bf16* __restrict__ out, int n) {
  int i = (blockIdx.x * 256 + threadIdx.x) * 4;
  if (i + 3 < n) {
    v4f v = *(const v4f*)(in + i);
    v4bf o;
    o.x = (bf16)v.x; o.y = (bf16)v.y; o.z = (bf16)v.z; o.w = (bf16)v.w;
    *(v4bf*)(out + i) = o;
  } else {
    for (int j = i; j < n; ++j) out[j] = (bf16)in[j];
  }
}

// ---------------------------------------------------------------- GEMM (NT)
// C[m,n] = sum_k A[m,k] * W[n,k];  A:[4096,1024] bf16 row-major, W:[1024,1024] bf16 row-major
// Double-buffered LDS tiles, async global->LDS fills overlapped with WMMA.
// MODE 0: out bf16, head layout [b,h,s,hd]
// MODE 1: out f32,  head layout [b,h,s,hd]
// MODE 2: out f32,  flat [m,n]
template <int MODE>
__global__ __launch_bounds__(256) void gemm_bf16_nt(
    const bf16* __restrict__ A, const bf16* __restrict__ W, void* __restrict__ outp) {
  constexpr int BM = 128, BN = 64, BK = 32, KDIM = 1024;
  __shared__ bf16 As[2][BM][BK + 8];  // +8 bf16 pad: rows 80B (16B aligned, bank-spread)
  __shared__ bf16 Bs[2][BN][BK + 8];

  const int tid  = threadIdx.x;
  const int lane = tid & 31;
  const int wv   = tid >> 5;   // wave 0..7
  const int wm   = wv >> 1;    // 0..3
  const int wn   = wv & 1;     // 0..1
  const int l15  = lane & 15;
  const int hiL  = lane >> 4;  // 0 or 1

  const int m0 = blockIdx.y * BM;
  const int n0 = blockIdx.x * BN;

  v8f acc[2][2];
  {
    v8f zero = {0, 0, 0, 0, 0, 0, 0, 0};
#pragma unroll
    for (int i = 0; i < 2; ++i)
#pragma unroll
      for (int j = 0; j < 2; ++j) acc[i][j] = zero;
  }

  // cooperative tile fill: 768 x 16B chunks across 256 threads
  auto load_tiles = [&](int k0, int buf) {
    int br = tid >> 2;
    int bc = (tid & 3) * 8;
    copy16_to_lds(W + (size_t)(n0 + br) * KDIM + k0 + bc, &Bs[buf][br][bc]);
#pragma unroll
    for (int p = 0; p < 2; ++p) {
      int ach = tid + p * 256;
      int ar  = ach >> 2;
      int ac  = (ach & 3) * 8;
      copy16_to_lds(A + (size_t)(m0 + ar) * KDIM + k0 + ac, &As[buf][ar][ac]);
    }
  };

  load_tiles(0, 0);
  WAIT_ASYNC();
  __syncthreads();

  int cur = 0;
  for (int k0 = 0; k0 < KDIM; k0 += BK) {
    const int nxt = cur ^ 1;
    if (k0 + BK < KDIM) load_tiles(k0 + BK, nxt);  // prefetch next tile (async)

    const int cA = hiL ? 8 : 0;    // A: lanes<16 hold K 0-7/16-23, lanes>=16 hold K 8-15/24-31
    const int cB = hiL ? 8 : 0;    // B: lanes<16 hold K 0-7/16-23, lanes>=16 hold K 8-15/24-31
    v16bf af[2], bfr[2];
#pragma unroll
    for (int i = 0; i < 2; ++i) {
      int r = wm * 32 + i * 16 + l15;
      ((v4f*)&af[i])[0] = *(const v4f*)(&As[cur][r][cA]);
      ((v4f*)&af[i])[1] = *(const v4f*)(&As[cur][r][cA + 16]);
    }
#pragma unroll
    for (int j = 0; j < 2; ++j) {
      int r = wn * 32 + j * 16 + l15;
      ((v4f*)&bfr[j])[0] = *(const v4f*)(&Bs[cur][r][cB]);
      ((v4f*)&bfr[j])[1] = *(const v4f*)(&Bs[cur][r][cB + 16]);
    }
#pragma unroll
    for (int i = 0; i < 2; ++i)
#pragma unroll
      for (int j = 0; j < 2; ++j) acc[i][j] = WMMA_BF16(af[i], bfr[j], acc[i][j]);

    WAIT_ASYNC();
    __syncthreads();
    cur = nxt;
  }

  // writeback: frag elem e -> row m = e + 8*hiL, col n = l15
#pragma unroll
  for (int i = 0; i < 2; ++i) {
#pragma unroll
    for (int j = 0; j < 2; ++j) {
      int n = n0 + wn * 32 + j * 16 + l15;
#pragma unroll
      for (int e = 0; e < 8; ++e) {
        int   m   = m0 + wm * 32 + i * 16 + e + hiL * 8;
        float val = acc[i][j][e];
        if (MODE == 2) {
          ((float*)outp)[(size_t)m * D_ + n] = val;
        } else {
          int    b   = m >> 11, s = m & (S_ - 1);
          int    h   = n >> 6, hd = n & (HD_ - 1);
          size_t off = ((size_t)(b * H_ + h) * S_ + s) * HD_ + hd;
          if (MODE == 0) ((bf16*)outp)[off] = (bf16)val;
          else           ((float*)outp)[off] = val;
        }
      }
    }
  }
}

// ---------------------------------------------------------------- sum of V rows per (b,h)
__global__ __launch_bounds__(64) void sumv_kernel(const float* __restrict__ Vf,
                                                  float* __restrict__ sumV) {
  int bh = blockIdx.x;
  int hd = threadIdx.x;
  const float* Vh = Vf + (size_t)bh * S_ * HD_;
  float a = 0.f;
  for (int s = 0; s < S_; ++s) a += Vh[(size_t)s * HD_ + hd];
  sumV[bh * HD_ + hd] = a;
}

// ---------------------------------------------------------------- attention (WMMA scores + top-3 + closed-form sparse softmax)
__global__ __launch_bounds__(256) void attn_topk_kernel(
    const bf16* __restrict__ Qb, const bf16* __restrict__ Kb,
    const float* __restrict__ Vf, const float* __restrict__ sumV,
    bf16* __restrict__ attnout) {
  constexpr int CHUNK = 768;  // 48 k-tiles, 48KB of f32 scores
  __shared__ float Sl[16][CHUNK];
  __shared__ float topv[16][3];
  __shared__ int   topi[16][3];
  __shared__ float candv[16][16][3];
  __shared__ int   candi[16][16][3];
  __shared__ float ez[16][4];
  __shared__ float sv[HD_];

  const int   qt   = blockIdx.x;   // q-tile (16 rows)
  const int   bh   = blockIdx.y;   // b*H + h
  const int   tid  = threadIdx.x;
  const int   lane = tid & 31;
  const int   wv   = tid >> 5;
  const int   l15  = lane & 15;
  const int   hiL  = lane >> 4;
  const float NEG_INF = -__builtin_inff();

  const bf16*  Qh = Qb + (size_t)bh * S_ * HD_;
  const bf16*  Kh = Kb + (size_t)bh * S_ * HD_;
  const float* Vh = Vf + (size_t)bh * S_ * HD_;

  if (tid < 16) {
    topv[tid][0] = NEG_INF; topv[tid][1] = NEG_INF; topv[tid][2] = NEG_INF;
    topi[tid][0] = 0; topi[tid][1] = 1; topi[tid][2] = 2;
  }
  if (tid < HD_) sv[tid] = sumV[bh * HD_ + tid];

  // A fragments (Q rows qt*16 .. qt*16+15), K-dim 64 => two 16x32 frags
  const bf16* Ap = Qh + (size_t)(qt * 16 + l15) * HD_;
  v16bf a0, a1;
  {
    int c = hiL ? 8 : 0;
    ((v4f*)&a0)[0] = *(const v4f*)(Ap + c);
    ((v4f*)&a0)[1] = *(const v4f*)(Ap + 16 + c);
    ((v4f*)&a1)[0] = *(const v4f*)(Ap + 32 + c);
    ((v4f*)&a1)[1] = *(const v4f*)(Ap + 48 + c);
  }
  __syncthreads();

  const int ntot = (qt + 1) * 16;  // causal: columns 0 .. qt*16+15
  for (int cb = 0; cb < ntot; cb += CHUNK) {
    const int ccols = (CHUNK < ntot - cb) ? CHUNK : (ntot - cb);
    const int tn    = ccols >> 4;
    for (int tt = wv; tt < tn; tt += 8) {
      const int   kt = (cb >> 4) + tt;
      const bf16* Kp = Kh + (size_t)(kt * 16 + l15) * HD_;
      v16bf b0, b1;
      int   c = hiL ? 8 : 0;
      ((v4f*)&b0)[0] = *(const v4f*)(Kp + c);
      ((v4f*)&b0)[1] = *(const v4f*)(Kp + c + 16);
      ((v4f*)&b1)[0] = *(const v4f*)(Kp + 32 + c);
      ((v4f*)&b1)[1] = *(const v4f*)(Kp + 32 + c + 16);
      v8f cf = {0, 0, 0, 0, 0, 0, 0, 0};
      cf = WMMA_BF16(a0, b0, cf);
      cf = WMMA_BF16(a1, b1, cf);
#pragma unroll
      for (int e = 0; e < 8; ++e) {
        int   m = e + hiL * 8;
        int   n = l15;
        float s = cf[e] * 0.125f;                 // 1/sqrt(64)
        if (kt == qt && n > m) s = NEG_INF;       // causal mask on diagonal tile
        Sl[m][tt * 16 + n] = s;
      }
    }
    __syncthreads();

    // per-row top-3 scan: 16 scanners per row, stride 16
    {
      int   r = tid >> 4, t = tid & 15;
      float v0 = NEG_INF, v1 = NEG_INF, v2 = NEG_INF;
      int   j0 = cb + t, j1 = cb + t, j2 = cb + t;
      for (int cc = t; cc < ccols; cc += 16) {
        float v = Sl[r][cc];
        int   g = cb + cc;
        if (v > v0)      { v2 = v1; j2 = j1; v1 = v0; j1 = j0; v0 = v; j0 = g; }
        else if (v > v1) { v2 = v1; j2 = j1; v1 = v;  j1 = g; }
        else if (v > v2) { v2 = v;  j2 = g; }
      }
      candv[r][t][0] = v0; candv[r][t][1] = v1; candv[r][t][2] = v2;
      candi[r][t][0] = j0; candi[r][t][1] = j1; candi[r][t][2] = j2;
    }
    __syncthreads();

    if ((tid & 15) == 0) {  // merge 16x3 candidates into running top-3
      int   r  = tid >> 4;
      float w0 = topv[r][0], w1 = topv[r][1], w2 = topv[r][2];
      int   u0 = topi[r][0], u1 = topi[r][1], u2 = topi[r][2];
      for (int t = 0; t < 16; ++t)
        for (int q = 0; q < 3; ++q) {
          float v = candv[r][t][q];
          int   g = candi[r][t][q];
          if (v > w0)      { w2 = w1; u2 = u1; w1 = w0; u1 = u0; w0 = v; u0 = g; }
          else if (v > w1) { w2 = w1; u2 = u1; w1 = v;  u1 = g; }
          else if (v > w2) { w2 = v;  u2 = g; }
        }
      topv[r][0] = w0; topv[r][1] = w1; topv[r][2] = w2;
      topi[r][0] = u0; topi[r][1] = u1; topi[r][2] = u2;
    }
    __syncthreads();
  }

  // sparse softmax over {3 selected scores, (S-3) zeros}:
  //   Z = (S-3) + sum_i exp(v_i);  out = (sumV + sum_i (exp(v_i)-1) * V[k_i]) / Z
  if (tid < 16) {
    int   r  = tid;
    float e0 = __expf(topv[r][0]);   // exp(-inf) == 0 handles short rows
    float e1 = __expf(topv[r][1]);
    float e2 = __expf(topv[r][2]);
    ez[r][0] = e0; ez[r][1] = e1; ez[r][2] = e2;
    ez[r][3] = (float)(S_ - TOPK_) + e0 + e1 + e2;
  }
  __syncthreads();

  {
    int r = tid >> 4, t = tid & 15;   // each thread: 4 hd values
    v4f acc = *(const v4f*)(&sv[t * 4]);
#pragma unroll
    for (int i = 0; i < 3; ++i) {
      float w  = ez[r][i] - 1.0f;
      v4f   vv = *(const v4f*)(Vh + (size_t)topi[r][i] * HD_ + t * 4);
      acc += w * vv;
    }
    float invZ = 1.0f / ez[r][3];
    int   b = bh >> 4, h = bh & 15;
    int   sg = qt * 16 + r;
    bf16* op = attnout + ((size_t)(b * S_ + sg) * D_ + h * HD_ + t * 4);
    op[0] = (bf16)(acc.x * invZ);
    op[1] = (bf16)(acc.y * invZ);
    op[2] = (bf16)(acc.z * invZ);
    op[3] = (bf16)(acc.w * invZ);
  }
}

// ---------------------------------------------------------------- launch
extern "C" void kernel_launch(void* const* d_in, const int* in_sizes, int n_in,
                              void* d_out, int out_size, void* d_ws, size_t ws_size,
                              hipStream_t stream) {
  const float* x  = (const float*)d_in[0];
  const float* wq = (const float*)d_in[1];
  const float* wk = (const float*)d_in[2];
  const float* wv = (const float*)d_in[3];
  const float* wo = (const float*)d_in[4];
  // d_in[5] = topk (== 3, baked in)
  (void)in_sizes; (void)n_in; (void)out_size; (void)ws_size;

  char*        ws = (char*)d_ws;
  const size_t MB = 1024 * 1024;
  bf16*  xb      = (bf16*)(ws + 0);        // 8 MB   x in bf16        [4096,1024]
  bf16*  wqb     = (bf16*)(ws + 8 * MB);   // 2 MB
  bf16*  wkb     = (bf16*)(ws + 10 * MB);  // 2 MB
  bf16*  wvb     = (bf16*)(ws + 12 * MB);  // 2 MB
  bf16*  wob     = (bf16*)(ws + 14 * MB);  // 2 MB
  bf16*  Qb      = (bf16*)(ws + 16 * MB);  // 8 MB   [b,h,s,hd] bf16
  bf16*  Kb      = (bf16*)(ws + 24 * MB);  // 8 MB   [b,h,s,hd] bf16
  float* Vf      = (float*)(ws + 32 * MB); // 16 MB  [b,h,s,hd] f32
  float* sumV    = (float*)(ws + 48 * MB); // 8 KB   [b*h, hd]
  bf16*  attnout = (bf16*)(ws + 49 * MB);  // 8 MB   [m, d] bf16

  auto cvt = [&](const float* in, bf16* out, int n) {
    int nb = (n / 4 + 255) / 256;
    f32_to_bf16_kernel<<<dim3(nb), dim3(256), 0, stream>>>(in, out, n);
  };
  cvt(x,  xb,  M_ * D_);
  cvt(wq, wqb, D_ * D_);
  cvt(wk, wkb, D_ * D_);
  cvt(wv, wvb, D_ * D_);
  cvt(wo, wob, D_ * D_);

  dim3 ggrid(D_ / 64, M_ / 128);  // (16, 32)
  gemm_bf16_nt<0><<<ggrid, dim3(256), 0, stream>>>(xb, wqb, (void*)Qb);
  gemm_bf16_nt<0><<<ggrid, dim3(256), 0, stream>>>(xb, wkb, (void*)Kb);
  gemm_bf16_nt<1><<<ggrid, dim3(256), 0, stream>>>(xb, wvb, (void*)Vf);

  sumv_kernel<<<dim3(B_ * H_), dim3(64), 0, stream>>>(Vf, sumV);

  attn_topk_kernel<<<dim3(S_ / 16, B_ * H_), dim3(256), 0, stream>>>(Qb, Kb, Vf, sumV, attnout);

  gemm_bf16_nt<2><<<ggrid, dim3(256), 0, stream>>>(attnout, wob, d_out);
}
